// HybridDnCNN_NoMultiScale_WithPreprocessing_14061722927877
// MI455X (gfx1250) — matrix operand order repaired and checked by reading, not compiled, not run on hardware
//
#include <hip/hip_runtime.h>
#include <hip/hip_bf16.h>

typedef _Float16 half_t;
typedef __attribute__((ext_vector_type(16))) _Float16 v16h;
typedef __attribute__((ext_vector_type(8)))  _Float16 v8h;
typedef __attribute__((ext_vector_type(8)))  float    v8f;
typedef __attribute__((ext_vector_type(4)))  unsigned int u32x4;
typedef __attribute__((ext_vector_type(8)))  int          i32x8;
typedef __attribute__((ext_vector_type(4)))  int          i32x4;

#define HH 256
#define WW 256
#define BB 8
#define CC 64
#define KTOT 576            // 64*9
#define NPIX (BB*HH*WW)
#define RAWSTR 34           // raw halo tile: [64][3][34], col fastest, NO padding (TDM-contiguous)
#define RAWROW 102          // 3*34
#define RAWSZ  (CC*RAWROW)  // 6528 halfs
#define BSTR   584          // ldsB row stride in halfs: 1168B, 16B aligned, conflict-free b128

#if defined(__has_builtin)
#if __has_builtin(__builtin_amdgcn_tensor_load_to_lds) && __has_builtin(__builtin_amdgcn_s_wait_tensorcnt)
#define HAVE_TDM 1
#endif
#endif

// ---------------------------------------------------------------- preprocess
__global__ void k_blend(const float* __restrict__ x, float* __restrict__ blended) {
    int idx = blockIdx.x * blockDim.x + threadIdx.x;
    if (idx >= NPIX) return;
    float v = (x[idx] + 1.0f) * 0.5f;
    v = fminf(fmaxf(v, 1e-6f), 1.0f);
    blended[idx] = 0.6f * v + 0.4f * powf(v, 1.5f);
}

__global__ void k_bilateral(const float* __restrict__ blended, float* __restrict__ xpre) {
    int idx = blockIdx.x * blockDim.x + threadIdx.x;
    if (idx >= NPIX) return;
    int b = idx >> 16, y = (idx >> 8) & 255, x = idx & 255;
    const float* img = blended + (b << 16);
    float center = img[(y << 8) + x];
    const float inv2sc = 1.0f / (2.0f * 0.05f * 0.05f);
    const float inv2ss = 1.0f / (2.0f * 50.0f * 50.0f);
    float num = 0.f, den = 0.f;
    #pragma unroll
    for (int dy = -2; dy <= 2; ++dy) {
        int yy = y + dy; yy = yy < 0 ? -yy : (yy > 255 ? 510 - yy : yy);  // reflect
        #pragma unroll
        for (int dx = -2; dx <= 2; ++dx) {
            int xx = x + dx; xx = xx < 0 ? -xx : (xx > 255 ? 510 - xx : xx);
            float nb = img[(yy << 8) + xx];
            float d  = nb - center;
            float w  = __expf(-(float)(dy*dy + dx*dx) * inv2ss) * __expf(-d*d*inv2sc);
            num += w * nb; den += w;
        }
    }
    xpre[idx] = num / den;
}

// ------------------------------------------------------------- weight cast
__global__ void k_cvt_f16(const float* __restrict__ src, half_t* __restrict__ dst, int n) {
    int idx = blockIdx.x * blockDim.x + threadIdx.x;
    if (idx < n) dst[idx] = (half_t)src[idx];
}

// ------------------------------------------------------------- init conv 1->64
__global__ void k_conv_init(const float* __restrict__ xpre, const float* __restrict__ w,
                            half_t* __restrict__ out) {
    int idx = blockIdx.x * blockDim.x + threadIdx.x;   // pixel
    if (idx >= NPIX) return;
    int b = idx >> 16, y = (idx >> 8) & 255, x = idx & 255;
    float v[9];
    #pragma unroll
    for (int ky = 0; ky < 3; ++ky)
        #pragma unroll
        for (int kx = 0; kx < 3; ++kx) {
            int gy = y + ky - 1, gx = x + kx - 1;
            v[ky*3+kx] = ((unsigned)gy < HH && (unsigned)gx < WW)
                         ? xpre[(b << 16) + (gy << 8) + gx] : 0.f;  // SAME zero pad
        }
    for (int co = 0; co < CC; ++co) {
        float acc = 0.f;
        #pragma unroll
        for (int t = 0; t < 9; ++t) acc = fmaf(w[co*9 + t], v[t], acc);
        acc = acc > 0.f ? acc : 0.f;  // relu
        out[((b*CC + co) * HH + y) * WW + x] = (half_t)acc;
    }
}

// -------------------------------------------------- body conv via WMMA GEMM
// GEMM: D[c_out(16) x pix(16)] += A[16x32 f16 weights] * B[32x16 f16 im2col]
// Block: 256 thr = 8 waves = 4 M-tiles x 2 pixel strips = 64 c_out x 32 pixels
// of one (b,y) row. Stage raw halo (TDM for interior blocks), then build
// pixel-major im2col in LDS with BN(prev)+ReLU folded in; K-loop feeds each
// WMMA with 2x ds_load_b128 (B) + 2x global b128 (A).
__global__ __launch_bounds__(256)
void k_conv_body(const half_t* __restrict__ in,         // [B][64][H][W] pre-BN (post-relu)
                 const half_t* __restrict__ w,          // [64][576] this layer, f16
                 const float*  __restrict__ scaleshift, // [0..63]=scale [64..127]=shift
                 half_t* __restrict__ out,              // [B][64][H][W]
                 float* __restrict__ gSum, float* __restrict__ gSumSq) {
    __shared__ half_t lds_raw[RAWSZ];          // raw halo: [c][ry][col], contiguous
    __shared__ half_t ldsB[32 * BSTR];         // im2col, pixel-major, K contiguous
    __shared__ float  lds_ss[2 * CC];          // scale | shift
    __shared__ float  lds_sum[CC], lds_sq[CC];

    const int tid = threadIdx.x;
    const int row = blockIdx.x;                // b*H + y
    const int b = row >> 8, y = row & 255;
    const int x0 = blockIdx.y * 32;

    if (tid < 2 * CC) lds_ss[tid] = scaleshift[tid];
    if (tid < CC) { lds_sum[tid] = 0.f; lds_sq[tid] = 0.f; }

    // ---- stage raw halo tile: [64ch][3 rows][34 cols] of f16 ----
    bool did_tdm = false;
#if HAVE_TDM
    const bool interior = (y >= 1) & (y <= HH - 2) & (blockIdx.y >= 1) & (blockIdx.y <= 6);
    if (interior) {
        if (tid < 32) {   // one wave issues one TDM DMA for the whole tile
            unsigned lds_off = (unsigned)(unsigned long long)(void*)lds_raw;
            unsigned long long ga = (unsigned long long)(const void*)
                (in + ((size_t)(b * CC) * HH + (y - 1)) * WW + (x0 - 1));
            // D# group0: count=1 | lds_addr | global_addr | type=2
            u32x4 g0 = { 1u, lds_off,
                         (unsigned)(ga & 0xFFFFFFFFu),
                         (unsigned)(((ga >> 32) & 0x01FFFFFFu) | 0x80000000u) };
            // D# group1: data_size=2B; tensor_dim0/1=65536; tile=34x3x64;
            //            dim0_stride=256, dim1_stride=65536
            i32x8 g1 = { (int)(1u << 16),          // wg_mask=0, data_size=1 (2B)
                         0,                         // ab_addr=0 | td0[15:0]=0
                         1,                         // td0[31:16]=1 | td1[15:0]=0
                         (int)((34u << 16) | 1u),   // td1[31:16]=1 | tile_dim0=34
                         (int)((64u << 16) | 3u),   // tile_dim1=3 | tile_dim2=64
                         256,                       // tensor_dim0_stride lo32
                         0,                         // stride0 hi | stride1 lo
                         1 };                       // tensor_dim1_stride hi32 (=65536)
            i32x4 g2 = { 0x10000, 0, 0, 0 };        // tensor_dim2=65536
            i32x4 g3 = { 0, 0, 0, 0 };
            i32x8 g4 = { 0, 0, 0, 0, 0, 0, 0, 0 };  // extra group (6-arg form), zero-filled
            __builtin_amdgcn_tensor_load_to_lds(g0, g1, g2, g3, g4, 0);
            __builtin_amdgcn_s_wait_tensorcnt(0);
        }
        did_tdm = true;
    }
#endif
    if (!did_tdm) {
        for (int idx = tid; idx < RAWSZ; idx += 256) {
            int c = idx / RAWROW, r = idx - c * RAWROW, ry = r / RAWSTR, col = r - ry * RAWSTR;
            int gy = y + ry - 1, gx = x0 + col - 1;
            lds_raw[idx] = ((unsigned)gy < HH && (unsigned)gx < WW)
                           ? in[((b*CC + c) * HH + gy) * WW + gx] : (half_t)0.f;
        }
    }
    __syncthreads();

    // ---- build pixel-major im2col with folded BN affine + relu ----
    // ldsB[p][k], k = c*9 + ky*3 + kx; OOB stays exactly 0 (true SAME pad).
    for (int id = tid; id < 32 * (KTOT / 8); id += 256) {
        int p = id & 31, kc8 = id >> 5;
        v8h chunk;
        #pragma unroll
        for (int j = 0; j < 8; ++j) {
            int k = kc8 * 8 + j;
            int c = k / 9, r = k - c * 9, ky = r / 3, kx = r - ky * 3;
            int gy = y + ky - 1, gx = x0 + p + kx - 1;
            float u = 0.f;
            if ((unsigned)gy < HH && (unsigned)gx < WW) {
                float v = (float)lds_raw[c * RAWROW + ky * RAWSTR + kx + p];
                v = fmaf(v, lds_ss[c], lds_ss[CC + c]);
                u = v > 0.f ? v : 0.f;
            }
            chunk[j] = (half_t)u;
        }
        *(v8h*)(ldsB + p * BSTR + kc8 * 8) = chunk;   // ds_store_b128
    }
    __syncthreads();

    // ---- WMMA K-loop ----
    const int wave = tid >> 5, lane = tid & 31;
    const int m_tile = wave & 3, strip = wave >> 2;
    const bool hi = lane >= 16;
    const int  M  = m_tile * 16 + (lane & 15);   // A row (c_out)
    const int  xl = strip * 16 + (lane & 15);    // B/C/D column (pixel)

    const half_t* wrow  = w + M * KTOT;
    const half_t* ldsBp = ldsB + xl * BSTR;
    const int bA   = hi ? 8 : 0;      // A layout: lo lanes K0-7/16-23, hi K8-15/24-31
    const int kadd = hi ? 8 : 0;      // B layout: lo lanes K0-7/16-23, hi K8-15/24-31

    v8f acc = {};
    #pragma unroll 3
    for (int kc = 0; kc < 18; ++kc) {
        const int kbase = kc * 32;
        __builtin_prefetch(wrow + kbase + 96, 0, 3);   // WGP-scope prefetch of next A chunk
        v16h a, bf;
        *(v8h*)&a        = *(const v8h*)(wrow + kbase + bA);        // global b128
        *((v8h*)&a + 1)  = *(const v8h*)(wrow + kbase + bA + 16);   // global b128
        *(v8h*)&bf       = *(const v8h*)(ldsBp + kbase + kadd);     // ds b128
        *((v8h*)&bf + 1) = *(const v8h*)(ldsBp + kbase + kadd + 16); // ds b128
        acc = __builtin_amdgcn_wmma_f32_16x16x32_f16(false, a, false, bf,
                                                     (short)0, acc, false, false);
    }

    // ---- ReLU, store f16 activation, BN stats via LDS f32 atomics ----
    // 32 lanes -> 2 addresses per ds_add_f32 (16-way) beats a shfl butterfly
    // (ds_bpermute + cndmask chains) by ~4x in issued ops.
    #pragma unroll
    for (int i = 0; i < 8; ++i) {
        float v = acc[i] > 0.f ? acc[i] : 0.f;
        int m = m_tile * 16 + i + (hi ? 8 : 0);   // C/D layout: row = vgpr + 8*laneHi
        out[((b*CC + m) * HH + y) * WW + x0 + xl] = (half_t)v;
        atomicAdd(&lds_sum[m], v);
        atomicAdd(&lds_sq[m],  v * v);
    }
    __syncthreads();
    if (tid < CC) {
        atomicAdd(&gSum[tid],   lds_sum[tid]);
        atomicAdd(&gSumSq[tid], lds_sq[tid]);
    }
}

// ------------------------------------------------------------- BN finalize
__global__ void k_bn_ident(float* scaleshift) {
    int i = threadIdx.x;
    if (i < 2 * CC) scaleshift[i] = (i < CC) ? 1.0f : 0.0f;
}

__global__ void k_bn_finalize(const float* __restrict__ gSum, const float* __restrict__ gSumSq,
                              const float* __restrict__ gamma, const float* __restrict__ beta,
                              float* __restrict__ scaleshift) {
    int c = threadIdx.x;
    if (c >= CC) return;
    const float n = (float)NPIX;
    float mean = gSum[c] / n;
    float var  = gSumSq[c] / n - mean * mean;
    float inv  = rsqrtf(var + 1e-5f);
    float s    = gamma[c] * inv;
    scaleshift[c]      = s;
    scaleshift[CC + c] = beta[c] - mean * s;
}

// --------------------------------------------------------- final conv 64->1
__global__ void k_conv_final(const half_t* __restrict__ h, const float* __restrict__ wfin,
                             const float* __restrict__ scaleshift,
                             const float* __restrict__ xpre, float* __restrict__ out) {
    int idx = blockIdx.x * blockDim.x + threadIdx.x;
    if (idx >= NPIX) return;
    int b = idx >> 16, y = (idx >> 8) & 255, x = idx & 255;
    const float* sc = scaleshift;
    const float* sh = scaleshift + CC;
    float acc = 0.f;
    for (int c = 0; c < CC; ++c) {
        const half_t* hp = h + ((size_t)(b*CC + c)) * HH * WW;
        float s = sc[c], t = sh[c];
        #pragma unroll
        for (int ky = 0; ky < 3; ++ky) {
            int gy = y + ky - 1; if ((unsigned)gy >= HH) continue;
            #pragma unroll
            for (int kx = 0; kx < 3; ++kx) {
                int gx = x + kx - 1; if ((unsigned)gx >= WW) continue;
                float v = (float)hp[(gy << 8) + gx];
                v = fmaf(v, s, t); v = v > 0.f ? v : 0.f;   // fold BN15 + relu
                acc = fmaf(wfin[(c*3 + ky)*3 + kx], v, acc);
            }
        }
    }
    out[idx] = xpre[idx] - acc;   // x_pre - noise
}

// ------------------------------------------------------------------- driver
extern "C" void kernel_launch(void* const* d_in, const int* in_sizes, int n_in,
                              void* d_out, int out_size, void* d_ws, size_t ws_size,
                              hipStream_t stream) {
    (void)in_sizes; (void)n_in; (void)out_size;
    const float* x        = (const float*)d_in[0];   // [8,1,256,256]
    const float* w_init   = (const float*)d_in[1];   // [64,1,3,3]
    const float* w_body   = (const float*)d_in[2];   // [15,64,64,3,3]
    const float* bn_gamma = (const float*)d_in[3];   // [15,64]
    const float* bn_beta  = (const float*)d_in[4];   // [15,64]
    const float* w_final  = (const float*)d_in[5];   // [1,64,3,3]
    float* out = (float*)d_out;

    char* ws = (char*)d_ws;
    size_t o = 0;
    auto carve = [&](size_t bytes) { char* p = ws + o; o = (o + bytes + 255) & ~(size_t)255; return p; };
    half_t* actA    = (half_t*)carve((size_t)BB * CC * HH * WW * 2);  // 64 MB
    half_t* actB    = (half_t*)carve((size_t)BB * CC * HH * WW * 2);  // 64 MB
    float*  blended = (float*) carve((size_t)NPIX * 4);
    float*  xpre    = (float*) carve((size_t)NPIX * 4);
    half_t* wb16    = (half_t*)carve((size_t)15 * CC * KTOT * 2);
    float*  stats   = (float*) carve(2 * CC * 4);   // gSum | gSumSq
    float*  ssbuf   = (float*) carve(2 * CC * 4);   // scale | shift
    if (o > ws_size) return;  // workspace too small; bail deterministically

    const int TB = 256;
    k_blend    <<<(NPIX + TB - 1) / TB, TB, 0, stream>>>(x, blended);
    k_bilateral<<<(NPIX + TB - 1) / TB, TB, 0, stream>>>(blended, xpre);
    k_cvt_f16<<<(15 * CC * KTOT + TB - 1) / TB, TB, 0, stream>>>(w_body, wb16, 15 * CC * KTOT);
    k_conv_init<<<(NPIX + TB - 1) / TB, TB, 0, stream>>>(xpre, w_init, actA);
    k_bn_ident<<<1, 128, 0, stream>>>(ssbuf);

    half_t* cur = actA; half_t* nxt = actB;
    dim3 grid(BB * HH, WW / 32);
    for (int l = 0; l < 15; ++l) {
        (void)hipMemsetAsync(stats, 0, 2 * CC * sizeof(float), stream);
        k_conv_body<<<grid, TB, 0, stream>>>(cur, wb16 + (size_t)l * CC * KTOT, ssbuf,
                                             nxt, stats, stats + CC);
        k_bn_finalize<<<1, CC, 0, stream>>>(stats, stats + CC,
                                            bn_gamma + l * CC, bn_beta + l * CC, ssbuf);
        half_t* t = cur; cur = nxt; nxt = t;
    }
    k_conv_final<<<(NPIX + TB - 1) / TB, TB, 0, stream>>>(cur, w_final, ssbuf, xpre, out);
}
